// GNN_80814104642188
// MI455X (gfx1250) — compile-verified, matrix operand order checked by reading, not run on hardware
//
#include <hip/hip_runtime.h>
#include <hip/hip_bf16.h>

typedef __attribute__((ext_vector_type(16))) _Float16 v16h;
typedef __attribute__((ext_vector_type(8)))  _Float16 v8h;
typedef __attribute__((ext_vector_type(4)))  _Float16 v4h;
typedef __attribute__((ext_vector_type(8)))  float    v8f;

// ---------------------------------------------------------------------------
// CDNA5 16-bit A/B fragment layout (16x16x32):
//   lanes 0-15 : elements 0-7 -> K=0-7,   elements 8-15 -> K=16-23
//   lanes 16-31: elements 0-7 -> K=8-15,  elements 8-15 -> K=24-31
// => each 8-element group is 8 CONSECUTIVE K values
// ---------------------------------------------------------------------------
__device__ __forceinline__ int wmma_klocal(int i, int lane) {
    int base = (i < 8) ? i : (16 + (i - 8));
    return base + ((lane & 16) ? 8 : 0);
}

// ---------------------------------------------------------------------------
// CSR construction: histogram -> LDS scan -> fill
// ---------------------------------------------------------------------------
__global__ void k_zero_i32(int* a, int n) {
    int i = blockIdx.x * blockDim.x + threadIdx.x;
    if (i < n) a[i] = 0;
}

__global__ void k_count_edges(const int* __restrict__ edst, int* __restrict__ cnt, int e) {
    int i = blockIdx.x * blockDim.x + threadIdx.x;
    if (i < e) atomicAdd(&cnt[edst[i]], 1);
}

__global__ void k_dinv_from_cnt(const int* __restrict__ cnt, float* __restrict__ dinv, int n) {
    int i = blockIdx.x * blockDim.x + threadIdx.x;
    if (i < n) dinv[i] = rsqrtf((float)(cnt[i] + 1));   // +1 self loop, always > 0
}

// per-block inclusive scan (Hillis-Steele in LDS), emit exclusive prefix + block sum
__global__ void k_scan1(const int* __restrict__ cnt, int* __restrict__ excl,
                        int* __restrict__ blockSums, int n) {
    __shared__ int sh[256];
    int tid = threadIdx.x;
    int i = blockIdx.x * 256 + tid;
    int v = (i < n) ? cnt[i] : 0;
    sh[tid] = v;
    __syncthreads();
#pragma unroll
    for (int off = 1; off < 256; off <<= 1) {
        int t = (tid >= off) ? sh[tid - off] : 0;
        __syncthreads();
        sh[tid] += t;
        __syncthreads();
    }
    if (i < n) excl[i] = sh[tid] - v;
    if (tid == 255) blockSums[blockIdx.x] = sh[255];
}

// single-block exclusive scan of block sums (nb <= 1024)
__global__ void k_scan2(int* blockSums, int nb) {
    __shared__ int sh[1024];
    int tid = threadIdx.x;
    int v = (tid < nb) ? blockSums[tid] : 0;
    sh[tid] = v;
    __syncthreads();
#pragma unroll
    for (int off = 1; off < 1024; off <<= 1) {
        int t = (tid >= off) ? sh[tid - off] : 0;
        __syncthreads();
        sh[tid] += t;
        __syncthreads();
    }
    if (tid < nb) blockSums[tid] = sh[tid] - v;
}

__global__ void k_scan3(int* __restrict__ excl, const int* __restrict__ blockSums, int n) {
    int i = blockIdx.x * 256 + threadIdx.x;
    if (i < n) excl[i] += blockSums[blockIdx.x];
}

__global__ void k_copy_i32(const int* __restrict__ src, int* __restrict__ dst, int n) {
    int i = blockIdx.x * blockDim.x + threadIdx.x;
    if (i < n) dst[i] = src[i];
}

// fill CSR slots: eidx[slot]=src, enorm[slot]=dinv[s]*dinv[d]; cursor ends as rowEnd
__global__ void k_fill_csr(const int* __restrict__ esrc, const int* __restrict__ edst,
                           const float* __restrict__ dinv, int* __restrict__ cursor,
                           int* __restrict__ eidx, float* __restrict__ enorm, int e) {
    int i = blockIdx.x * blockDim.x + threadIdx.x;
    if (i >= e) return;
    int s = esrc[i];
    int d = edst[i];
    int slot = atomicAdd(&cursor[d], 1);
    eidx[slot] = s;
    enorm[slot] = dinv[s] * dinv[d];
}

// ---------------------------------------------------------------------------
// Weight pre-pack kernels: fragment order [tileN][kChunk][lane][16 halves]
// ---------------------------------------------------------------------------
template <int IC, int OC>
__global__ void k_pack_conv_w(const float* __restrict__ w, _Float16* __restrict__ out,
                              int total) {
    int idx = blockIdx.x * blockDim.x + threadIdx.x;
    if (idx >= total) return;
    constexpr int KTOT = 9 * IC;
    constexpr int KC = (KTOT + 31) / 32;
    int i    = idx & 15;
    int lane = (idx >> 4) & 31;
    int rest = idx >> 9;             // nT*KC + kc
    int kc   = rest % KC;
    int nT   = rest / KC;
    int k    = kc * 32 + wmma_klocal(i, lane);
    int oc   = nT * 16 + (lane & 15);
    float v = 0.0f;
    if (k < KTOT && oc < OC) {
        int t = k / IC;
        int c = k - t * IC;
        v = w[((long long)oc * IC + c) * 9 + t];   // OIHW, tap-major K
    }
    out[idx] = (_Float16)v;
}

__global__ void k_pack_w64(const float* __restrict__ W, _Float16* __restrict__ out) {
    int idx = blockIdx.x * blockDim.x + threadIdx.x;   // 4 colT * 2 kc * 32 * 16 = 4096
    if (idx >= 4096) return;
    int i    = idx & 15;
    int lane = (idx >> 4) & 31;
    int kc   = (idx >> 9) & 1;
    int colT = idx >> 10;
    int k    = kc * 32 + wmma_klocal(i, lane);
    int col  = colT * 16 + (lane & 15);
    out[idx] = (_Float16)W[k * 64 + col];
}

// ---------------------------------------------------------------------------
// GCN layer 1: tiny GEMM [N,4] @ [4,64] (K=4 -> plain VALU)
// ---------------------------------------------------------------------------
__global__ void k_gemm_k4(const float* __restrict__ x, const float* __restrict__ W,
                          float* __restrict__ out, int n) {
    int idx = blockIdx.x * blockDim.x + threadIdx.x;
    if (idx >= n * 64) return;
    int p = idx >> 6;
    int o = idx & 63;
    float acc = 0.0f;
#pragma unroll
    for (int k = 0; k < 4; ++k) acc = fmaf(x[p * 4 + k], W[k * 64 + o], acc);
    out[idx] = acc;
}

// ---------------------------------------------------------------------------
// GCN layers 2/3: [N,64]f16 @ packed W -> [N,64]f32 via v_wmma_f32_16x16x32_f16
// ---------------------------------------------------------------------------
__global__ void k_gemm64_wmma(const _Float16* __restrict__ h,
                              const _Float16* __restrict__ Wpk,
                              float* __restrict__ out) {
    const int lane   = threadIdx.x & 31;
    const int wave   = threadIdx.x >> 5;   // column tile 0..3
    const int rowT   = blockIdx.x;
    const int m      = lane & 15;
    const int row    = rowT * 16 + m;
    const int laneHi = (lane & 16) ? 8 : 0;
    const _Float16* arow  = h + (long long)row * 64 + laneHi;
    const _Float16* bfrag = Wpk + ((long long)wave * 2 * 32 + lane) * 16;

    v8f acc = {};
#pragma unroll
    for (int kc = 0; kc < 2; ++kc) {
        v16h a;
        v8h lo = *(const v8h*)(arow + kc * 32);
        v8h hi = *(const v8h*)(arow + kc * 32 + 16);
#pragma unroll
        for (int i = 0; i < 8; ++i) { a[i] = lo[i]; a[8 + i] = hi[i]; }
        v16h b = *(const v16h*)(bfrag + (long long)kc * 32 * 16);
        acc = __builtin_amdgcn_wmma_f32_16x16x32_f16(false, a, false, b,
                                                     (short)0, acc, false, false);
    }

#pragma unroll
    for (int r = 0; r < 8; ++r) {
        int rr = rowT * 16 + r + laneHi;
        out[(long long)rr * 64 + wave * 16 + m] = acc[r];
    }
}

// ---------------------------------------------------------------------------
// Fused CSR gather + self-loop + bias + relu + f32->f16 store.
// 16 threads per node, 4 channels each. tmp is L2-resident (67MB < 192MB L2).
// ---------------------------------------------------------------------------
__global__ void k_gather_relu_f16(const float* __restrict__ tmp, const float* __restrict__ dinv,
                                  const int* __restrict__ rowStart, const int* __restrict__ rowEnd,
                                  const int* __restrict__ eidx, const float* __restrict__ enorm,
                                  const float* __restrict__ bias, _Float16* __restrict__ out,
                                  int n) {
    long long tid = (long long)blockIdx.x * blockDim.x + threadIdx.x;
    int node = (int)(tid >> 4);
    if (node >= n) return;
    int c0 = (int)(tid & 15) * 4;

    float di = dinv[node];
    const float4 self = *(const float4*)(tmp + (long long)node * 64 + c0);
    const float4 bb   = *(const float4*)(bias + c0);
    float4 acc;
    acc.x = self.x * di * di + bb.x;
    acc.y = self.y * di * di + bb.y;
    acc.z = self.z * di * di + bb.z;
    acc.w = self.w * di * di + bb.w;

    int j0 = rowStart[node], j1 = rowEnd[node];
    for (int j = j0; j < j1; ++j) {
        int s   = eidx[j];
        float w = enorm[j];
        const float4 v = *(const float4*)(tmp + (long long)s * 64 + c0);
        acc.x = fmaf(v.x, w, acc.x);
        acc.y = fmaf(v.y, w, acc.y);
        acc.z = fmaf(v.z, w, acc.z);
        acc.w = fmaf(v.w, w, acc.w);
    }

    v4h o;
    o[0] = (_Float16)fmaxf(acc.x, 0.0f);
    o[1] = (_Float16)fmaxf(acc.y, 0.0f);
    o[2] = (_Float16)fmaxf(acc.z, 0.0f);
    o[3] = (_Float16)fmaxf(acc.w, 0.0f);
    *(v4h*)(out + (long long)node * 64 + c0) = o;
}

// ---------------------------------------------------------------------------
// Conv 3x3 SAME as implicit GEMM via WMMA (NHWC f16 in, packed f16 weights).
// ---------------------------------------------------------------------------
template <int IC, int OC, bool RELU, bool TO_CHW>
__global__ void k_conv3x3_wmma(const _Float16* __restrict__ in,
                               const _Float16* __restrict__ wpk,
                               const float* __restrict__ bias,
                               void* __restrict__ outp, int G) {
    const int lane = threadIdx.x & 31;
    const int wave = threadIdx.x >> 5;
    const int tile = blockIdx.x * 4 + wave;          // pixel tile (16 px of one row)
    const int tilesPerRow = G >> 4;
    const int py  = tile / tilesPerRow;
    const int px0 = (tile - py * tilesPerRow) << 4;
    const int nT  = blockIdx.y;
    const int m   = lane & 15;
    const int oc  = nT * 16 + m;
    constexpr int KTOT = 9 * IC;
    constexpr int KC = (KTOT + 31) / 32;
    const int laneHi = (lane & 16) ? 8 : 0;

    const _Float16* bfrag = wpk + ((long long)nT * KC * 32 + lane) * 16;

    v8f acc = {};
    for (int kc = 0; kc < KC; ++kc) {
        v16h a;
#pragma unroll
        for (int g = 0; g < 2; ++g) {
            int kbase = kc * 32 + g * 16 + laneHi;   // 8 consecutive K values
            v8h av = {};
            if (kbase < KTOT) {
                int t  = kbase / IC;                 // IC power of two -> shift
                int c0 = kbase - t * IC;
                int dy = t / 3 - 1;
                int dx = t % 3 - 1;
                int yy = py + dy;
                int xx = px0 + m + dx;
                if (yy >= 0 && yy < G && xx >= 0 && xx < G)
                    av = *(const v8h*)(in + ((long long)yy * G + xx) * IC + c0);
            }
#pragma unroll
            for (int i = 0; i < 8; ++i) a[g * 8 + i] = av[i];
        }
        v16h b = *(const v16h*)(bfrag + (long long)kc * 32 * 16);
        acc = __builtin_amdgcn_wmma_f32_16x16x32_f16(false, a, false, b,
                                                     (short)0, acc, false, false);
    }

    float bval = (oc < OC) ? bias[oc] : 0.0f;
#pragma unroll
    for (int r = 0; r < 8; ++r) {
        long long pixel = (long long)py * G + px0 + r + laneHi;
        float v = acc[r] + bval;
        if (RELU) v = fmaxf(v, 0.0f);
        if (oc < OC) {
            if (TO_CHW)
                ((float*)outp)[(long long)oc * G * G + pixel] = v;
            else
                ((_Float16*)outp)[pixel * OC + oc] = (_Float16)v;
        }
    }
}

// ---------------------------------------------------------------------------
// Host side
// ---------------------------------------------------------------------------
extern "C" void kernel_launch(void* const* d_in, const int* in_sizes, int n_in,
                              void* d_out, int out_size, void* d_ws, size_t ws_size,
                              hipStream_t stream) {
    const float* x    = (const float*)d_in[0];
    const int*   esrc = (const int*)d_in[1];
    const int*   edst = (const int*)d_in[2];
    const float* W1 = (const float*)d_in[4];
    const float* b1 = (const float*)d_in[5];
    const float* W2 = (const float*)d_in[6];
    const float* b2 = (const float*)d_in[7];
    const float* W3 = (const float*)d_in[8];
    const float* b3 = (const float*)d_in[9];
    const float* cw1 = (const float*)d_in[10];
    const float* cb1 = (const float*)d_in[11];
    const float* cw2 = (const float*)d_in[12];
    const float* cb2 = (const float*)d_in[13];
    const float* cw3 = (const float*)d_in[14];
    const float* cb3 = (const float*)d_in[15];
    const float* cw4 = (const float*)d_in[16];
    const float* cb4 = (const float*)d_in[17];

    const int N = in_sizes[0] / 4;
    const int E = in_sizes[1];
    int G = 1;
    while ((long long)G * G < N) G++;

    // ---- workspace layout (bytes) ----
    char* p = (char*)d_ws;
    int*   cnt      = (int*)p;          p += (size_t)N * 4;
    int*   rowStart = (int*)p;          p += (size_t)N * 4;
    int*   cursor   = (int*)p;          p += (size_t)N * 4;
    int*   blockSums= (int*)p;          p += 4096;
    float* dinv     = (float*)p;        p += (size_t)N * 4;
    float* tmp      = (float*)p;        p += (size_t)N * 256;   // N*64 f32
    _Float16* hF16  = (_Float16*)p;     p += (size_t)N * 128;   // N*64 f16
    int*   eidx     = (int*)p;          p += (size_t)E * 4;
    float* enorm    = (float*)p;        p += (size_t)E * 4;
    _Float16* pW2   = (_Float16*)p;     p += 4096 * 2;
    _Float16* pW3   = (_Float16*)p;     p += 4096 * 2;
    const int PC1 = 2 * 18 * 512, PC2 = 1 * 9 * 512, PC3 = 1 * 5 * 512, PC4 = 1 * 3 * 512;
    _Float16* pc1 = (_Float16*)p;       p += (size_t)PC1 * 2;
    _Float16* pc2 = (_Float16*)p;       p += (size_t)PC2 * 2;
    _Float16* pc3 = (_Float16*)p;       p += (size_t)PC3 * 2;
    _Float16* pc4 = (_Float16*)p;       p += (size_t)PC4 * 2;
    _Float16* c1out = (_Float16*)p;     p += (size_t)N * 32 * 2;
    _Float16* c2out = (_Float16*)p;     p += (size_t)N * 16 * 2;
    _Float16* c3out = (_Float16*)p;     p += (size_t)N * 8 * 2;

    const int T = 256;
    const long long NE = (long long)N * 64;
    const int elemBlocks = (int)((NE + T - 1) / T);
    const int nodeBlocks = (N + T - 1) / T;
    const int edgeBlocks = (E + T - 1) / T;
    const int gemmBlocks = N / 16;
    const int scanBlocks = (N + 255) / 256;   // 1024 for N=512*512 (fits k_scan2)
    const int gatherBlocks = (int)(((long long)N * 16 + T - 1) / T);

    // ---- weight packing (small, one pass) ----
    k_pack_w64<<<(4096 + T - 1) / T, T, 0, stream>>>(W2, pW2);
    k_pack_w64<<<(4096 + T - 1) / T, T, 0, stream>>>(W3, pW3);
    k_pack_conv_w<64, 32><<<(PC1 + T - 1) / T, T, 0, stream>>>(cw1, pc1, PC1);
    k_pack_conv_w<32, 16><<<(PC2 + T - 1) / T, T, 0, stream>>>(cw2, pc2, PC2);
    k_pack_conv_w<16, 8 ><<<(PC3 + T - 1) / T, T, 0, stream>>>(cw3, pc3, PC3);
    k_pack_conv_w<8,  6 ><<<(PC4 + T - 1) / T, T, 0, stream>>>(cw4, pc4, PC4);

    // ---- CSR build: histogram -> scan -> fill ----
    k_zero_i32<<<nodeBlocks, T, 0, stream>>>(cnt, N);
    k_count_edges<<<edgeBlocks, T, 0, stream>>>(edst, cnt, E);
    k_dinv_from_cnt<<<nodeBlocks, T, 0, stream>>>(cnt, dinv, N);
    k_scan1<<<scanBlocks, 256, 0, stream>>>(cnt, rowStart, blockSums, N);
    k_scan2<<<1, 1024, 0, stream>>>(blockSums, scanBlocks);
    k_scan3<<<scanBlocks, 256, 0, stream>>>(rowStart, blockSums, N);
    k_copy_i32<<<nodeBlocks, T, 0, stream>>>(rowStart, cursor, N);
    k_fill_csr<<<edgeBlocks, T, 0, stream>>>(esrc, edst, dinv, cursor, eidx, enorm, E);
    // cursor now holds rowEnd

    // ---- GCN layer 1 ----
    k_gemm_k4<<<elemBlocks, T, 0, stream>>>(x, W1, tmp, N);
    k_gather_relu_f16<<<gatherBlocks, T, 0, stream>>>(tmp, dinv, rowStart, cursor,
                                                      eidx, enorm, b1, hF16, N);
    // ---- GCN layer 2 ----
    k_gemm64_wmma<<<gemmBlocks, 128, 0, stream>>>(hF16, pW2, tmp);
    k_gather_relu_f16<<<gatherBlocks, T, 0, stream>>>(tmp, dinv, rowStart, cursor,
                                                      eidx, enorm, b2, hF16, N);
    // ---- GCN layer 3 ----
    k_gemm64_wmma<<<gemmBlocks, 128, 0, stream>>>(hF16, pW3, tmp);
    k_gather_relu_f16<<<gatherBlocks, T, 0, stream>>>(tmp, dinv, rowStart, cursor,
                                                      eidx, enorm, b3, hF16, N);

    // ---- CNN stack (f16 NHWC ping-pong; final layer f32 CHW to d_out) ----
    const int pixTiles = (G * G) / 16;
    dim3 convGrid1(pixTiles / 4, 2);
    dim3 convGrid2(pixTiles / 4, 1);
    dim3 convGrid3(pixTiles / 4, 1);
    dim3 convGrid4(pixTiles / 4, 1);

    k_conv3x3_wmma<64, 32, true,  false><<<convGrid1, 128, 0, stream>>>(hF16,  pc1, cb1, (void*)c1out, G);
    k_conv3x3_wmma<32, 16, true,  false><<<convGrid2, 128, 0, stream>>>(c1out, pc2, cb2, (void*)c2out, G);
    k_conv3x3_wmma<16, 8,  true,  false><<<convGrid3, 128, 0, stream>>>(c2out, pc3, cb3, (void*)c3out, G);
    k_conv3x3_wmma<8,  6,  false, true ><<<convGrid4, 128, 0, stream>>>(c3out, pc4, cb4, d_out, G);
}
